// APDIntelligibilityEstimator_14912126451945
// MI455X (gfx1250) — matrix operand order repaired and checked by reading, not compiled, not run on hardware
//
#include <hip/hip_runtime.h>
#include <math.h>
#include <stdint.h>

// ---------------------------------------------------------------------------
// Problem constants (from reference setup_inputs)
// ---------------------------------------------------------------------------
#define BATCH   16
#define TIN     64000
#define TDIM    3201          // (64000 + 2*20 - 40)/20 + 1
#define CDIM    512           // E and C
#define BNDIM   256
#define HDIM    256
#define MROWS   (BATCH * TDIM)   // 51216, multiple of 16
#define N_TCN   24
#define EPSV    1e-5f

typedef unsigned int   u32;
typedef unsigned short u16;

// ---------------------------------------------------------------------------
// bf16 helpers (storage as unsigned short; RNE conversion)
// ---------------------------------------------------------------------------
__device__ __forceinline__ u16 f2bf(float f) {
    u32 u = __float_as_uint(f);
    u = (u + 0x7FFFu + ((u >> 16) & 1u)) >> 16;
    return (u16)u;
}
__device__ __forceinline__ float bf2f(u16 h) {
    return __uint_as_float(((u32)h) << 16);
}

// WMMA vector types (probe-confirmed signatures)
typedef __attribute__((ext_vector_type(16))) __bf16 bf16x16;
typedef __attribute__((ext_vector_type(8)))  float  f32x8;

// ---------------------------------------------------------------------------
// CDNA5 async memory->LDS copy (ASYNCcnt path) + wait helper
//   global_load_async_to_lds_b128 VDST(lds byte addr), VADDR(64b global), off
// ---------------------------------------------------------------------------
__device__ __forceinline__ void async_copy_b128(const void* g, void* lds) {
    u32 l = (u32)(uintptr_t)lds;   // generic->LDS: low 32 bits are LDS byte addr
    asm volatile("global_load_async_to_lds_b128 %0, %1, off"
                 :: "v"(l), "v"(g)
                 : "memory");
}

template<int N>
__device__ __forceinline__ void wait_asynccnt() {
#if __has_builtin(__builtin_amdgcn_s_wait_asynccnt)
    __builtin_amdgcn_s_wait_asynccnt(N);
#else
    asm volatile("s_wait_asynccnt %0" :: "i"(N) : "memory");
#endif
}

// ---------------------------------------------------------------------------
// Encoder conv: (B,1,64000) -> (B,T,512) f32, k=40 s=20 p=20
// ---------------------------------------------------------------------------
__global__ __launch_bounds__(256) void enc_conv_kernel(
    const float* __restrict__ x, const float* __restrict__ w,
    float* __restrict__ out)
{
    size_t flat = (size_t)blockIdx.x * 256 + threadIdx.x;   // < MROWS*512
    int e = (int)(flat & 511u);
    size_t bt = flat >> 9;
    int t = (int)(bt % TDIM);
    int b = (int)(bt / TDIM);
    const float* xr = x + (size_t)b * TIN;
    const float* wr = w + e * 40;
    int base = t * 20 - 20;
    float acc = 0.f;
    #pragma unroll
    for (int k = 0; k < 40; ++k) {
        int ix = base + k;
        if (ix >= 0 && ix < TIN) acc = fmaf(xr[ix], wr[k], acc);
    }
    out[flat] = acc;
}

// ---------------------------------------------------------------------------
// Per-batch sum / sum-of-squares reduction (for GroupNorm(1))
// One block = 256 contiguous elements => single batch per block.
// ---------------------------------------------------------------------------
__global__ __launch_bounds__(256) void stats_reduce_kernel(
    const float* __restrict__ in, float* __restrict__ stats, int perB)
{
    __shared__ float red[512];
    int tid = threadIdx.x;
    size_t idx = (size_t)blockIdx.x * 256 + tid;
    float v = in[idx];
    int b = (int)(idx / (size_t)perB);
    red[tid] = v; red[tid + 256] = v * v;
    __syncthreads();
    for (int off = 128; off; off >>= 1) {
        if (tid < off) { red[tid] += red[tid + off]; red[tid + 256] += red[tid + 256 + off]; }
        __syncthreads();
    }
    if (tid == 0) {
        atomicAdd(&stats[2 * b + 0], red[0]);
        atomicAdd(&stats[2 * b + 1], red[256]);
    }
}

// ---------------------------------------------------------------------------
// GroupNorm(1) + PReLU + quantize to bf16  (C = 512 innermost)
// ---------------------------------------------------------------------------
__global__ __launch_bounds__(256) void gn_prelu_bf16_kernel(
    const float* __restrict__ in, u16* __restrict__ outh,
    const float* __restrict__ stats, const float* __restrict__ gamma,
    const float* __restrict__ beta, const float* __restrict__ prelu_a,
    int perB, float invCnt)
{
    size_t idx = (size_t)blockIdx.x * 256 + threadIdx.x;
    int c = (int)(idx & 511u);
    int b = (int)(idx / (size_t)perB);
    float mean = stats[2 * b + 0] * invCnt;
    float var  = stats[2 * b + 1] * invCnt - mean * mean;
    float rstd = rsqrtf(var + EPSV);
    float v = (in[idx] - mean) * rstd * gamma[c] + beta[c];
    float a = prelu_a[0];
    v = (v >= 0.f) ? v : a * v;
    outh[idx] = f2bf(v);
}

// ---------------------------------------------------------------------------
// Depthwise dilated conv k=3 (pad=d) + GroupNorm stats, layout (B,T,C)
// ---------------------------------------------------------------------------
__global__ __launch_bounds__(256) void dwconv_stats_kernel(
    const float* __restrict__ in, const float* __restrict__ w3, int d,
    float* __restrict__ out, float* __restrict__ stats)
{
    __shared__ float red[512];
    int tid = threadIdx.x;
    size_t idx = (size_t)blockIdx.x * 256 + tid;
    int c = (int)(idx & 511u);
    size_t bt = idx >> 9;
    int t = (int)(bt % TDIM);
    int b = (int)(bt / TDIM);
    float w0 = w3[c * 3 + 0], w1 = w3[c * 3 + 1], w2 = w3[c * 3 + 2];
    float acc = w1 * in[idx];
    if (t >= d)        acc = fmaf(w0, in[idx - (size_t)d * CDIM], acc);
    if (t + d < TDIM)  acc = fmaf(w2, in[idx + (size_t)d * CDIM], acc);
    out[idx] = acc;
    red[tid] = acc; red[tid + 256] = acc * acc;
    __syncthreads();
    for (int off = 128; off; off >>= 1) {
        if (tid < off) { red[tid] += red[tid + off]; red[tid + 256] += red[tid + 256 + off]; }
        __syncthreads();
    }
    if (tid == 0) {
        atomicAdd(&stats[2 * b + 0], red[0]);
        atomicAdd(&stats[2 * b + 1], red[256]);
    }
}

// ---------------------------------------------------------------------------
// Weight prep: |w| sum per slot (for w_scale = mean|w|) and sign->bf16
// ---------------------------------------------------------------------------
__global__ __launch_bounds__(256) void abs_sum_kernel(
    const float* __restrict__ src, int perSlot, float* __restrict__ dst)
{
    __shared__ float red[256];
    int tid = threadIdx.x;
    size_t idx = (size_t)blockIdx.x * 256 + tid;
    red[tid] = fabsf(src[idx]);
    __syncthreads();
    for (int off = 128; off; off >>= 1) {
        if (tid < off) red[tid] += red[tid + off];
        __syncthreads();
    }
    if (tid == 0) atomicAdd(&dst[(size_t)blockIdx.x * 256 / (size_t)perSlot], red[0]);
}

__global__ __launch_bounds__(256) void sign_bf16_kernel(
    const float* __restrict__ src, u16* __restrict__ dst, int n)
{
    int i = blockIdx.x * 256 + threadIdx.x;
    if (i < n) {
        float w = src[i];
        float s = (w > 0.f) ? 1.f : ((w < 0.f) ? -1.f : 0.f);
        dst[i] = f2bf(s);   // exact
    }
}

// ---------------------------------------------------------------------------
// bf16 WMMA GEMM:  out(M,N) = A(M,K) * Bw(N,K)^T * alpha [+ resid]
//   alpha = (wsum * inv_wcnt) * scale[0]    (binarized-weight scaling)
// Block: 256 threads = 8 waves as 2(M) x 4(N); block tile 128x128; K-step 32.
// Each wave: 4 M-frags x 2 N-frags of v_wmma_f32_16x16x32_bf16.
// Double-buffered LDS fed by global_load_async_to_lds_b128 (ASYNCcnt),
// 4 async instructions per wave per stage -> s_wait_asynccnt 4 pipelining.
// LDS rows padded to 40 halves (80B) to avoid ds bank conflicts.
// ---------------------------------------------------------------------------
#define LDT 40

__global__ __launch_bounds__(256) void gemm_bsign_kernel(
    const u16* __restrict__ A, const u16* __restrict__ Bw,
    const float* __restrict__ resid, float* __restrict__ outF,
    u16* __restrict__ outH,
    const float* __restrict__ wsum, float inv_wcnt,
    const float* __restrict__ scale,
    int M, int N, int K)
{
    __shared__ u16 As[2][128 * LDT];
    __shared__ u16 Bs[2][128 * LDT];

    const int tid  = threadIdx.x;
    const int lane = tid & 31;
    const int wave = tid >> 5;
    const int wm   = wave >> 2;      // 0..1  -> 64 rows of M
    const int wn   = wave & 3;       // 0..3  -> 32 cols of N
    const int l16  = lane & 15;
    const int lh   = lane >> 4;      // lane group (ISA 16-bit operand layout)
    const int m0   = blockIdx.y * 128;
    const int n0   = blockIdx.x * 128;

    // per-thread staging coordinates: 512 chunks of 16B per tile, 2 per thread
    const int row_c = tid >> 2;            // 0..63  (+64 for second chunk)
    const int cc    = (tid & 3) * 8;       // element offset within 32-wide row
    int gmA0 = m0 + row_c;       gmA0 = (gmA0 < M) ? gmA0 : (M - 1);
    int gmA1 = m0 + row_c + 64;  gmA1 = (gmA1 < M) ? gmA1 : (M - 1);
    const int gnB0 = n0 + row_c;           // N is a multiple of 128 here
    const int gnB1 = n0 + row_c + 64;
    const u16* gA0 = A  + (size_t)gmA0 * K + cc;
    const u16* gA1 = A  + (size_t)gmA1 * K + cc;
    const u16* gB0 = Bw + (size_t)gnB0 * K + cc;
    const u16* gB1 = Bw + (size_t)gnB1 * K + cc;

    // issue one stage of async copies (4 instructions per wave)
    auto stage = [&](int buf, int k0) {
        async_copy_b128(gA0 + k0, &As[buf][ row_c       * LDT + cc]);
        async_copy_b128(gA1 + k0, &As[buf][(row_c + 64) * LDT + cc]);
        async_copy_b128(gB0 + k0, &Bs[buf][ row_c       * LDT + cc]);
        async_copy_b128(gB1 + k0, &Bs[buf][(row_c + 64) * LDT + cc]);
    };

    f32x8 acc[4][2] = {};

    const int nk = K >> 5;        // K-tiles of 32
    stage(0, 0);
    for (int kt = 0; kt < nk; ++kt) {
        const int buf = kt & 1;
        if (kt + 1 < nk) {
            stage(buf ^ 1, (kt + 1) << 5);   // prefetch next tile into alt buffer
            wait_asynccnt<4>();              // our previous-stage copies complete
        } else {
            wait_asynccnt<0>();
        }
        __syncthreads();                     // all waves' copies visible in LDS

        // ---- build fragments per CDNA5 16-bit WMMA operand layout ----
        bf16x16 afrag[4], bfrag[2];
        #pragma unroll
        for (int i = 0; i < 4; ++i) {
            int r = wm * 64 + i * 16 + l16;                 // A row (M)
            union { uint4 u[2]; bf16x16 v; } uu;
            uu.u[0] = *(const uint4*)(&As[buf][r * LDT + lh * 8]);       // K 0-7 / 8-15
            uu.u[1] = *(const uint4*)(&As[buf][r * LDT + lh * 8 + 16]);  // K 16-23 / 24-31
            afrag[i] = uu.v;
        }
        #pragma unroll
        for (int j = 0; j < 2; ++j) {
            int r = wn * 32 + j * 16 + l16;                 // B row (N); Bs is [N][K]
            union { uint4 u[2]; bf16x16 v; } uu;
            uu.u[0] = *(const uint4*)(&Bs[buf][r * LDT + lh * 8]);       // K 0-7 / 8-15
            uu.u[1] = *(const uint4*)(&Bs[buf][r * LDT + lh * 8 + 16]);  // K 16-23 / 24-31
            bfrag[j] = uu.v;
        }

        #pragma unroll
        for (int i = 0; i < 4; ++i)
            #pragma unroll
            for (int j = 0; j < 2; ++j)
                acc[i][j] = __builtin_amdgcn_wmma_f32_16x16x32_bf16(
                    false, afrag[i], false, bfrag[j],
                    (short)0, acc[i][j], false, false);

        __syncthreads();   // readers done before next stage overwrites this buffer
    }

    // ---- epilogue: scale, residual, f32 and/or bf16 store ----
    const float alpha = wsum[0] * inv_wcnt * scale[0];
    #pragma unroll
    for (int i = 0; i < 4; ++i) {
        #pragma unroll
        for (int j = 0; j < 2; ++j) {
            #pragma unroll
            for (int r = 0; r < 8; ++r) {
                int m = m0 + wm * 64 + i * 16 + lh * 8 + r;   // C/D layout: VGPR r, lane group
                int n = n0 + wn * 32 + j * 16 + l16;
                if (m < M) {
                    size_t idx = (size_t)m * N + n;
                    float v = acc[i][j][r] * alpha;
                    if (resid) v += resid[idx];
                    if (outF)  outF[idx] = v;
                    if (outH)  outH[idx] = f2bf(v);
                }
            }
        }
    }
}

// ---------------------------------------------------------------------------
// Mean over T: (B,T,512) -> (B,512)
// ---------------------------------------------------------------------------
__global__ __launch_bounds__(256) void mean_t_kernel(
    const float* __restrict__ in, float* __restrict__ h)
{
    int idx = blockIdx.x * 256 + threadIdx.x;   // < B*512
    int c = idx & 511, b = idx >> 9;
    const float* p = in + (size_t)b * TDIM * CDIM + c;
    float s = 0.f;
    for (int t = 0; t < TDIM; ++t) s += p[(size_t)t * CDIM];
    h[idx] = s * (1.f / (float)TDIM);
}

// ---------------------------------------------------------------------------
// Head: bit_linear(512->256) + PReLU + bit_linear(256->1) + sigmoid.
// x_scale does NOT cancel here (bias term), so compute it faithfully.
// One block, 256 threads.
// ---------------------------------------------------------------------------
__global__ __launch_bounds__(256) void head_kernel(
    const float* __restrict__ h,
    const float* __restrict__ fc1_w, const float* __restrict__ fc1_b,
    const float* __restrict__ fc1_scale, const float* __restrict__ head_prelu,
    const float* __restrict__ fco_w, const float* __restrict__ fco_b,
    const float* __restrict__ fco_scale, const float* __restrict__ wsums,
    float* __restrict__ out)
{
    __shared__ float sh[BATCH * 512];
    __shared__ float sh2[BATCH * HDIM];
    __shared__ float xs[BATCH];
    __shared__ float xs2[BATCH];
    int tid = threadIdx.x;

    for (int i = tid; i < BATCH * 512; i += 256) sh[i] = h[i];
    __syncthreads();
    if (tid < BATCH) {
        float s = 0.f;
        for (int k = 0; k < 512; ++k) s += fabsf(sh[tid * 512 + k]);
        s *= (1.f / 512.f);
        xs[tid] = (s > EPSV) ? s : EPSV;
    }
    __syncthreads();

    const float ws1 = wsums[26] * (1.f / (256.f * 512.f));
    const float sc1 = fc1_scale[0];
    const float pr  = head_prelu[0];
    // thread tid owns output feature j = tid (0..255)
    for (int b = 0; b < BATCH; ++b) {
        float dot = 0.f;
        for (int k = 0; k < 512; ++k) {
            float w = fc1_w[tid * 512 + k];
            float sw = (w > 0.f) ? 1.f : ((w < 0.f) ? -1.f : 0.f);
            dot = fmaf(sh[b * 512 + k], sw, dot);
        }
        float v = (dot / xs[b] + fc1_b[tid]) * ws1 * xs[b] * sc1;
        v = (v >= 0.f) ? v : pr * v;
        sh2[b * HDIM + tid] = v;
    }
    __syncthreads();
    if (tid < BATCH) {
        float s = 0.f;
        for (int k = 0; k < HDIM; ++k) s += fabsf(sh2[tid * HDIM + k]);
        s *= (1.f / (float)HDIM);
        xs2[tid] = (s > EPSV) ? s : EPSV;
    }
    __syncthreads();
    if (tid < BATCH) {
        const float ws2 = wsums[27] * (1.f / (float)HDIM);
        float dot = 0.f;
        for (int k = 0; k < HDIM; ++k) {
            float w = fco_w[k];
            float sw = (w > 0.f) ? 1.f : ((w < 0.f) ? -1.f : 0.f);
            dot = fmaf(sh2[tid * HDIM + k], sw, dot);
        }
        float v = (dot / xs2[tid] + fco_b[0]) * ws2 * xs2[tid] * fco_scale[0];
        out[tid] = 1.f / (1.f + expf(-v));
    }
}

// ---------------------------------------------------------------------------
// Orchestration
// ---------------------------------------------------------------------------
extern "C" void kernel_launch(void* const* d_in, const int* in_sizes, int n_in,
                              void* d_out, int out_size, void* d_ws, size_t ws_size,
                              hipStream_t stream)
{
    (void)in_sizes; (void)n_in; (void)out_size; (void)ws_size;

    const float* x         = (const float*)d_in[0];
    const float* enc_w     = (const float*)d_in[1];
    const float* enc_gamma = (const float*)d_in[2];
    const float* enc_beta  = (const float*)d_in[3];
    const float* enc_prelu = (const float*)d_in[4];
    const float* bn_w      = (const float*)d_in[5];
    const float* bn_scale  = (const float*)d_in[6];
    const float* ti_w      = (const float*)d_in[7];
    const float* ti_scale  = (const float*)d_in[8];
    const float* dw_w      = (const float*)d_in[9];
    const float* pw_w      = (const float*)d_in[10];
    const float* pw_scale  = (const float*)d_in[11];
    const float* tcn_gamma = (const float*)d_in[12];
    const float* tcn_beta  = (const float*)d_in[13];
    const float* tcn_prelu = (const float*)d_in[14];
    const float* fc1_w     = (const float*)d_in[15];
    const float* fc1_b     = (const float*)d_in[16];
    const float* fc1_scale = (const float*)d_in[17];
    const float* head_prelu= (const float*)d_in[18];
    const float* fco_w     = (const float*)d_in[19];
    const float* fco_b     = (const float*)d_in[20];
    const float* fco_scale = (const float*)d_in[21];
    float* out = (float*)d_out;

    // ---- carve workspace (256B aligned) ----
    char* p = (char*)d_ws;
    auto carve = [&](size_t bytes) -> void* {
        void* r = (void*)p;
        p += (bytes + 255) & ~(size_t)255;
        return r;
    };
    float* buf0    = (float*)carve((size_t)MROWS * CDIM * 4);   // f32 activations
    float* buf1    = (float*)carve((size_t)MROWS * CDIM * 4);   // dwconv tmp
    u16*   bufh    = (u16*)  carve((size_t)MROWS * CDIM * 2);   // bf16 GEMM input (K=512)
    u16*   bufh2   = (u16*)  carve((size_t)MROWS * BNDIM * 2);  // bf16 GEMM input (K=256)
    u16*   wsgn_bn = (u16*)  carve((size_t)BNDIM * CDIM * 2);
    u16*   wsgn_ti = (u16*)  carve((size_t)CDIM * BNDIM * 2);
    u16*   wsgn_pw = (u16*)  carve((size_t)N_TCN * CDIM * CDIM * 2);
    float* wsums   = (float*)carve(28 * 4);    // 0=bn 1=ti 2..25=pw 26=fc1 27=fco
    float* stats   = (float*)carve(2 * BATCH * 4);
    float* hbuf    = (float*)carve((size_t)BATCH * CDIM * 4);

    const int ELEMS   = MROWS * CDIM;          // 26,222,592
    const int EBLOCKS = ELEMS / 256;           // 102,432 (divides exactly)
    const int PERB    = TDIM * CDIM;           // 1,638,912
    const float INVCNT = 1.f / (float)PERB;

    // ---- weight prep ----
    hipMemsetAsync(wsums, 0, 28 * 4, stream);
    abs_sum_kernel<<<(BNDIM * CDIM) / 256, 256, 0, stream>>>(bn_w, BNDIM * CDIM, wsums + 0);
    abs_sum_kernel<<<(CDIM * BNDIM) / 256, 256, 0, stream>>>(ti_w, CDIM * BNDIM, wsums + 1);
    abs_sum_kernel<<<(N_TCN * CDIM * CDIM) / 256, 256, 0, stream>>>(pw_w, CDIM * CDIM, wsums + 2);
    abs_sum_kernel<<<(HDIM * CDIM) / 256, 256, 0, stream>>>(fc1_w, HDIM * CDIM, wsums + 26);
    abs_sum_kernel<<<1, 256, 0, stream>>>(fco_w, HDIM, wsums + 27);
    sign_bf16_kernel<<<(BNDIM * CDIM) / 256, 256, 0, stream>>>(bn_w, wsgn_bn, BNDIM * CDIM);
    sign_bf16_kernel<<<(CDIM * BNDIM) / 256, 256, 0, stream>>>(ti_w, wsgn_ti, CDIM * BNDIM);
    sign_bf16_kernel<<<(N_TCN * CDIM * CDIM) / 256, 256, 0, stream>>>(pw_w, wsgn_pw, N_TCN * CDIM * CDIM);

    // ---- encoder: conv -> GroupNorm(1) -> PReLU -> bf16 ----
    enc_conv_kernel<<<EBLOCKS, 256, 0, stream>>>(x, enc_w, buf0);
    hipMemsetAsync(stats, 0, 2 * BATCH * 4, stream);
    stats_reduce_kernel<<<EBLOCKS, 256, 0, stream>>>(buf0, stats, PERB);
    gn_prelu_bf16_kernel<<<EBLOCKS, 256, 0, stream>>>(buf0, bufh, stats, enc_gamma,
                                                      enc_beta, enc_prelu, PERB, INVCNT);

    const dim3 gN2(2, (MROWS + 127) / 128);   // N = 256
    const dim3 gN4(4, (MROWS + 127) / 128);   // N = 512

    // ---- bottleneck (512->256) and tcn_input (256->512) bit convs (x_scale cancels) ----
    gemm_bsign_kernel<<<gN2, 256, 0, stream>>>(bufh, wsgn_bn, nullptr, nullptr, bufh2,
                                               wsums + 0, 1.f / (float)(BNDIM * CDIM),
                                               bn_scale, MROWS, BNDIM, CDIM);
    gemm_bsign_kernel<<<gN4, 256, 0, stream>>>(bufh2, wsgn_ti, nullptr, buf0, nullptr,
                                               wsums + 1, 1.f / (float)(CDIM * BNDIM),
                                               ti_scale, MROWS, CDIM, BNDIM);

    // ---- 24 TCN layers ----
    for (int i = 0; i < N_TCN; ++i) {
        int d = 1 << (i % 8);
        hipMemsetAsync(stats, 0, 2 * BATCH * 4, stream);
        dwconv_stats_kernel<<<EBLOCKS, 256, 0, stream>>>(buf0, dw_w + (size_t)i * CDIM * 3,
                                                         d, buf1, stats);
        gn_prelu_bf16_kernel<<<EBLOCKS, 256, 0, stream>>>(buf1, bufh, stats,
                                                          tcn_gamma + (size_t)i * CDIM,
                                                          tcn_beta + (size_t)i * CDIM,
                                                          tcn_prelu + i, PERB, INVCNT);
        gemm_bsign_kernel<<<gN4, 256, 0, stream>>>(bufh, wsgn_pw + (size_t)i * CDIM * CDIM,
                                                   buf0, buf0, nullptr,
                                                   wsums + 2 + i, 1.f / (float)(CDIM * CDIM),
                                                   pw_scale + i, MROWS, CDIM, CDIM);
    }

    // ---- head ----
    mean_t_kernel<<<(BATCH * CDIM) / 256, 256, 0, stream>>>(buf0, hbuf);
    head_kernel<<<1, 256, 0, stream>>>(hbuf, fc1_w, fc1_b, fc1_scale, head_prelu,
                                       fco_w, fco_b, fco_scale, wsums, out);
}
